// CaMoE_System_75831942578809
// MI455X (gfx1250) — matrix operand order repaired and checked by reading, not compiled, not run on hardware
//
#include <hip/hip_runtime.h>
#include <hip/hip_bf16.h>
#include <math.h>

// ---------------- model dims ----------------
#define DB   4
#define DT   1024
#define DC   1024
#define DV   32000
#define DL   2
#define DNR  6
#define DNT  2
#define DE   8
#define DHID 4096
#define DP   8
#define DR   64
#define DNH  16
#define DHD  64
#define NROWS (DB*DT)            // 4096

// ---------------- WMMA types ----------------
typedef __attribute__((ext_vector_type(16))) __bf16 v16bf;
typedef __attribute__((ext_vector_type(8)))  float  v8f;
typedef __attribute__((ext_vector_type(4)))  unsigned short us4;

__device__ __forceinline__ unsigned short f32_to_bf16_raw(float f) {
    union { float f; unsigned u; } a; a.f = f;
    unsigned r = a.u + 0x7FFFu + ((a.u >> 16) & 1u);   // round-to-nearest-even
    return (unsigned short)(r >> 16);
}
__device__ __forceinline__ __bf16 raw_to_bf16(unsigned short u) {
    __bf16 b; __builtin_memcpy(&b, &u, 2); return b;
}
__device__ __forceinline__ float dsigmoid(float x) { return 1.0f / (1.0f + __expf(-x)); }

// =====================================================================
// GEMM: C[M,N] (+)= act(A[M,K] * B[K,N]) * row_scale[m]
// f32 in/out, bf16 WMMA core (v_wmma_f32_16x16x32_bf16), f32 accumulate.
// Block: 256 thr = 8 wave32 (2x4). Block tile 128x128, K-step 32.
// Wave tile 64x32 = 4x2 WMMA 16x16 accumulators.
// Software-pipelined: next K-chunk fetched into registers during compute.
// REQUIRES (all call sites satisfy): M % 128 == 0, K % 32 == 0, N % 4 == 0.
// act: 0=none 1=sigmoid 2=squared-relu
// =====================================================================
#define BM 128
#define BN 128
#define BK 32
#define PADK 40   // BK+8 ushorts = 80B row stride: 16B aligned, bank-spread

__global__ __launch_bounds__(256)
void k_gemm(const float* __restrict__ A, const float* __restrict__ B,
            float* __restrict__ Cd, int M, int N, int K,
            int lda, int ldb, int ldc,
            int act, const float* __restrict__ row_scale, int scale_stride,
            int accum)
{
    __shared__ unsigned short As [BM][PADK];   // A row-major  [m][k]
    __shared__ unsigned short BsT[BN][PADK];   // B transposed [n][k]

    const int tid   = threadIdx.x;
    const int bN0   = blockIdx.x * BN;
    const int bM0   = blockIdx.y * BM;
    const int wave  = tid >> 5;
    const int lane  = tid & 31;
    const int wm    = wave & 1;          // 2 waves along M
    const int wn    = wave >> 1;         // 4 waves along N
    const int lrow  = lane & 15;
    const int khalf = lane >> 4;

    v8f acc[4][2] = {};
    float4 fa[4], fb[4];

    // load one K-chunk of A/B tiles into registers (loads issue back-to-back)
    auto load_regs = [&](int kk) {
        #pragma unroll
        for (int i = 0; i < 4; ++i) {
            int vid = tid + i * 256;
            int rr  = vid >> 3;            // A row 0..127
            int vc  = (vid & 7) * 4;       // A col 0,4,...,28
            fa[i] = *reinterpret_cast<const float4*>(
                A + (size_t)(bM0 + rr) * lda + (kk + vc));
        }
        #pragma unroll
        for (int i = 0; i < 4; ++i) {
            int vid = tid + i * 256;
            int rr  = vid >> 5;            // B k-row 0..31
            int vc  = (vid & 31) * 4;      // B n-col 0,4,...,124
            int gn  = bN0 + vc;
            fb[i] = make_float4(0.f, 0.f, 0.f, 0.f);
            if (gn < N)                     // N % 4 == 0 -> whole-vec guard
                fb[i] = *reinterpret_cast<const float4*>(
                    B + (size_t)(kk + rr) * ldb + gn);
        }
    };
    // convert staged registers to bf16 and store to LDS
    auto store_lds = [&]() {
        #pragma unroll
        for (int i = 0; i < 4; ++i) {
            int vid = tid + i * 256;
            int rr  = vid >> 3;
            int vc  = (vid & 7) * 4;
            us4 u;
            u[0] = f32_to_bf16_raw(fa[i].x);
            u[1] = f32_to_bf16_raw(fa[i].y);
            u[2] = f32_to_bf16_raw(fa[i].z);
            u[3] = f32_to_bf16_raw(fa[i].w);
            *reinterpret_cast<us4*>(&As[rr][vc]) = u;   // ds_store_b64
        }
        #pragma unroll
        for (int i = 0; i < 4; ++i) {
            int vid = tid + i * 256;
            int rr  = vid >> 5;
            int vc  = (vid & 31) * 4;
            BsT[vc + 0][rr] = f32_to_bf16_raw(fb[i].x); // transposed scatter
            BsT[vc + 1][rr] = f32_to_bf16_raw(fb[i].y);
            BsT[vc + 2][rr] = f32_to_bf16_raw(fb[i].z);
            BsT[vc + 3][rr] = f32_to_bf16_raw(fb[i].w);
        }
    };

    load_regs(0);                           // pipeline prologue

    for (int k0 = 0; k0 < K; k0 += BK) {
        store_lds();
        __syncthreads();

        // issue next chunk's global loads early: latency hides under WMMAs
        if (k0 + BK < K) {
            load_regs(k0 + BK);
            if (k0 + 2 * BK < K && tid < BM)    // warm L2 two chunks ahead
                __builtin_prefetch(&A[(size_t)(bM0 + tid) * lda + (k0 + 2 * BK)], 0, 1);
        }

        // ---- gather fragments in contiguous k order (k = 16*khalf + e), same for A and B ----
        v16bf afrag[4];
        #pragma unroll
        for (int t = 0; t < 4; ++t) {
            int m = wm * 64 + t * 16 + lrow;
            #pragma unroll
            for (int e = 0; e < 16; ++e) {
                int kk = khalf * 16 + e;                        // A 16x32, contiguous k (matches B)
                afrag[t][e] = raw_to_bf16(As[m][kk]);
            }
        }
        v16bf bfrag[2];
        #pragma unroll
        for (int s = 0; s < 2; ++s) {
            int n = wn * 32 + s * 16 + lrow;
            #pragma unroll
            for (int e = 0; e < 16; ++e) {
                int kk = khalf * 16 + e;                        // B 32x16, contiguous k (matches A)
                bfrag[s][e] = raw_to_bf16(BsT[n][kk]);          // contiguous in LDS
            }
        }
        // ---- 8 WMMAs per wave per K-chunk ----
        #pragma unroll
        for (int t = 0; t < 4; ++t)
            #pragma unroll
            for (int s = 0; s < 2; ++s)
                acc[t][s] = __builtin_amdgcn_wmma_f32_16x16x32_bf16(
                    false, afrag[t], false, bfrag[s], (short)0, acc[t][s], false, false);

        __syncthreads();
    }

    // ---- epilogue: C/D layout m = r + 8*khalf, n = lane&15 ----
    #pragma unroll
    for (int t = 0; t < 4; ++t) {
        #pragma unroll
        for (int s = 0; s < 2; ++s) {
            int n = bN0 + wn * 32 + s * 16 + lrow;
            if (n < N) {
                #pragma unroll
                for (int r = 0; r < 8; ++r) {
                    int m = bM0 + wm * 64 + t * 16 + r + khalf * 8;
                    float v = acc[t][s][r];
                    if (act == 1)      v = dsigmoid(v);
                    else if (act == 2) { float rl = v > 0.0f ? v : 0.0f; v = rl * rl; }
                    if (row_scale)     v *= row_scale[(size_t)m * scale_stride];
                    size_t o = (size_t)m * ldc + n;
                    if (accum) Cd[o] += v; else Cd[o] = v;
                }
            }
        }
    }
}

// =====================================================================
// small support kernels
// =====================================================================
__global__ void k_embed(const int* __restrict__ idx, const float* __restrict__ emb,
                        float* __restrict__ x) {
    size_t i = (size_t)blockIdx.x * blockDim.x + threadIdx.x;
    if (i >= (size_t)NROWS * DC) return;
    int n = (int)(i / DC), c = (int)(i % DC);
    x[i] = emb[(size_t)idx[n] * DC + c];
}

__global__ void k_ln(const float* __restrict__ x, const float* __restrict__ g,
                     const float* __restrict__ b, float* __restrict__ y) {
    int n = blockIdx.x, tid = threadIdx.x;
    const float* row = x + (size_t)n * DC;
    float s = 0.f, s2 = 0.f;
    for (int c = tid; c < DC; c += 256) { float v = row[c]; s += v; s2 += v * v; }
    for (int o = 16; o > 0; o >>= 1) { s += __shfl_down(s, o, 32); s2 += __shfl_down(s2, o, 32); }
    __shared__ float ss[8], ss2[8];
    __shared__ float mn_s, rs_s;
    int wv = tid >> 5, ln = tid & 31;
    if (ln == 0) { ss[wv] = s; ss2[wv] = s2; }
    __syncthreads();
    if (tid == 0) {
        float ts = 0.f, ts2 = 0.f;
        for (int w = 0; w < 8; ++w) { ts += ss[w]; ts2 += ss2[w]; }
        float mean = ts / DC;
        float var  = ts2 / DC - mean * mean;
        mn_s = mean; rs_s = rsqrtf(var + 1e-5f);
    }
    __syncthreads();
    float mean = mn_s, rstd = rs_s;
    float* out = y + (size_t)n * DC;
    for (int c = tid; c < DC; c += 256) out[c] = (row[c] - mean) * rstd * g[c] + b[c];
}

// per-channel RWKV scan: s = w*s + k*v over T
__global__ void k_scan(const float* __restrict__ k, const float* __restrict__ v,
                       const float* __restrict__ decay, float* __restrict__ st) {
    int ch = blockIdx.x * blockDim.x + threadIdx.x;   // over B*C
    if (ch >= DB * DC) return;
    int b = ch / DC, c = ch % DC;
    float w = dsigmoid(decay[c]);
    float s = 0.f;
    size_t base = (size_t)b * DT * DC + c;
    for (int t = 0; t < DT; ++t) {
        size_t o = base + (size_t)t * DC;
        s = w * s + k[o] * v[o];
        st[o] = s;
    }
}

__global__ void k_copy(float* __restrict__ d, const float* __restrict__ s, size_t n) {
    size_t i = (size_t)blockIdx.x * blockDim.x + threadIdx.x;
    if (i < n) d[i] = s[i];
}
__global__ void k_mix(float* __restrict__ v, const float* __restrict__ vf, size_t n) {
    size_t i = (size_t)blockIdx.x * blockDim.x + threadIdx.x;
    if (i < n) v[i] = 0.5f * (v[i] + vf[i]);
}
__global__ void k_mul(float* __restrict__ o, const float* __restrict__ a,
                      const float* __restrict__ b, size_t n) {
    size_t i = (size_t)blockIdx.x * blockDim.x + threadIdx.x;
    if (i < n) o[i] = a[i] * b[i];
}
__global__ void k_add(float* __restrict__ x, const float* __restrict__ y, size_t n) {
    size_t i = (size_t)blockIdx.x * blockDim.x + threadIdx.x;
    if (i < n) x[i] += y[i];
}
__global__ void k_add2(float* __restrict__ o, const float* __restrict__ a,
                       const float* __restrict__ b, size_t n) {
    size_t i = (size_t)blockIdx.x * blockDim.x + threadIdx.x;
    if (i < n) o[i] = a[i] + b[i];
}
__global__ void k_zero(float* __restrict__ p, size_t n) {
    size_t i = (size_t)blockIdx.x * blockDim.x + threadIdx.x;
    if (i < n) p[i] = 0.f;
}

// routing: one block (256 thr) per row -> gate[n][E]
__global__ void k_routing(const float* __restrict__ h, const float* __restrict__ confr,
                          const float* __restrict__ conft, const float* __restrict__ wd,
                          const float* __restrict__ Wa, const float* __restrict__ capital,
                          float* __restrict__ gate) {
    int n = blockIdx.x, tid = threadIdx.x;
    const float* row = h + (size_t)n * DC;
    float vals[17];
    #pragma unroll
    for (int q = 0; q < 17; ++q) vals[q] = 0.f;
    for (int c = tid; c < DC; c += 256) {
        float hv = row[c];
        #pragma unroll
        for (int e = 0; e < DNR; ++e) vals[e] += hv * confr[e * DC + c];
        vals[6] += hv * conft[c];
        vals[7] += hv * conft[DC + c];
        vals[8] += hv * wd[c];
        #pragma unroll
        for (int e = 0; e < DE; ++e) vals[9 + e] += hv * Wa[(size_t)c * DE + e];
    }
    #pragma unroll
    for (int q = 0; q < 17; ++q)
        for (int o = 16; o > 0; o >>= 1) vals[q] += __shfl_down(vals[q], o, 32);
    __shared__ float sm[17][8];
    int wv = tid >> 5, ln = tid & 31;
    if (ln == 0) {
        #pragma unroll
        for (int q = 0; q < 17; ++q) sm[q][wv] = vals[q];
    }
    __syncthreads();
    if (tid == 0) {
        float tot[17];
        #pragma unroll
        for (int q = 0; q < 17; ++q) {
            float t = 0.f;
            for (int w = 0; w < 8; ++w) t += sm[q][w];
            tot[q] = t;
        }
        float dif = (tot[8] > 20.f) ? tot[8] : log1pf(__expf(tot[8]));  // softplus
        float bid[DE];
        #pragma unroll
        for (int e = 0; e < DE; ++e)
            bid[e] = dsigmoid(tot[e]) * capital[e] * dif + tanhf(tot[9 + e]);
        int i0 = 0;
        for (int e = 1; e < DE; ++e) if (bid[e] > bid[i0]) i0 = e;
        int i1 = (i0 == 0) ? 1 : 0;
        for (int e = 0; e < DE; ++e) if (e != i0 && bid[e] > bid[i1]) i1 = e;
        float ex = __expf(bid[i1] - bid[i0]);
        float w0 = 1.f / (1.f + ex), w1 = ex / (1.f + ex);
        for (int e = 0; e < DE; ++e)
            gate[(size_t)n * DE + e] = (e == i0) ? w0 : ((e == i1) ? w1 : 0.f);
    }
}

// prefix attention: grid (N, NH), block = HD (64). softmax over P=8
__global__ void k_attn(const float* __restrict__ q, const float* __restrict__ kp,
                       const float* __restrict__ vp, float* __restrict__ o) {
    int n = blockIdx.x, hh = blockIdx.y, d = threadIdx.x;
    __shared__ float red[DHD];
    __shared__ float sc[DP];
    const int off = hh * DHD + d;
    float qd = q[(size_t)n * DC + off];
    for (int p = 0; p < DP; ++p) {
        red[d] = qd * kp[(size_t)(n * DP + p) * DC + off];
        __syncthreads();
        for (int s = 32; s > 0; s >>= 1) { if (d < s) red[d] += red[d + s]; __syncthreads(); }
        if (d == 0) sc[p] = red[0] * 0.125f;   // 1/sqrt(64)
        __syncthreads();
    }
    float m = sc[0];
    #pragma unroll
    for (int p = 1; p < DP; ++p) m = fmaxf(m, sc[p]);
    float den = 0.f, w[DP];
    #pragma unroll
    for (int p = 0; p < DP; ++p) { w[p] = __expf(sc[p] - m); den += w[p]; }
    float acc = 0.f;
    #pragma unroll
    for (int p = 0; p < DP; ++p) acc += w[p] * vp[(size_t)(n * DP + p) * DC + off];
    o[(size_t)n * DC + off] = acc / den;
}

// =====================================================================
// host orchestration
// =====================================================================
static inline void launch_gemm(hipStream_t st, const float* A, const float* B, float* Cd,
                               int M, int N, int K, int lda, int ldb, int ldc,
                               int act, const float* rs, int rss, int accum) {
    dim3 grid((N + BN - 1) / BN, (M + BM - 1) / BM);
    k_gemm<<<grid, 256, 0, st>>>(A, B, Cd, M, N, K, lda, ldb, ldc, act, rs, rss, accum);
}

extern "C" void kernel_launch(void* const* d_in, const int* in_sizes, int n_in,
                              void* d_out, int out_size, void* d_ws, size_t ws_size,
                              hipStream_t stream) {
    const int*   idx       = (const int*)  d_in[0];
    const float* capital   = (const float*)d_in[1];
    const float* emb       = (const float*)d_in[2];
    const float* headW     = (const float*)d_in[3];
    const float* ln1_g     = (const float*)d_in[4];
    const float* ln1_b     = (const float*)d_in[5];
    const float* ln2_g     = (const float*)d_in[6];
    const float* ln2_b     = (const float*)d_in[7];
    const float* lno_g     = (const float*)d_in[8];
    const float* lno_b     = (const float*)d_in[9];
    const float* att_Wr    = (const float*)d_in[10];
    const float* att_Wk    = (const float*)d_in[11];
    const float* att_Wv    = (const float*)d_in[12];
    const float* att_Wo    = (const float*)d_in[13];
    const float* att_decay = (const float*)d_in[14];
    const float* critic_wd = (const float*)d_in[15];
    const float* critic_Wa = (const float*)d_in[16];
    const float* Wdown     = (const float*)d_in[17];
    const float* Wup       = (const float*)d_in[18];
    const float* eWk       = (const float*)d_in[19];
    const float* eWv       = (const float*)d_in[20];
    const float* conf_r    = (const float*)d_in[21];
    const float* tWq       = (const float*)d_in[22];
    const float* tWk       = (const float*)d_in[23];
    const float* tWv       = (const float*)d_in[24];
    const float* tWo       = (const float*)d_in[25];
    const float* conf_t    = (const float*)d_in[26];

    // ---- carve workspace ----
    float* w = (float*)d_ws;
    size_t off = 0;
    auto alloc = [&](size_t n) { float* p = w + off; off += n; return p; };
    const size_t NC = (size_t)NROWS * DC;
    float* x      = alloc(NC);
    float* h      = alloc(NC);
    float* rbuf   = alloc(NC);
    float* kbuf   = alloc(NC);
    float* vbuf   = alloc(NC);
    float* vfirst = alloc(NC);
    float* state  = alloc(NC);
    float* rsbuf  = alloc(NC);
    float* tmp    = alloc(NC);          // att_out / attention output
    float* qbuf   = alloc(NC);
    float* finalb = alloc(NC);
    float* atmp   = alloc(NC);
    float* gate   = alloc((size_t)NROWS * DE);
    float* ubuf   = alloc((size_t)NROWS * DR);
    float* prefix = alloc((size_t)NROWS * DP * DC);
    float* kp     = alloc((size_t)NROWS * DP * DC);
    float* vp     = alloc((size_t)NROWS * DP * DC);
    float* hid    = alloc((size_t)NROWS * DHID);
    (void)ws_size; (void)in_sizes; (void)n_in; (void)out_size;

    const int ebN  = (int)((NC + 255) / 256);

    // ---- embedding ----
    k_embed<<<ebN, 256, 0, stream>>>(idx, emb, x);

    for (int l = 0; l < DL; ++l) {
        const float* Wr = att_Wr + (size_t)l * DC * DC;
        const float* Wk = att_Wk + (size_t)l * DC * DC;
        const float* Wv = att_Wv + (size_t)l * DC * DC;
        const float* Wo = att_Wo + (size_t)l * DC * DC;

        // ---- TimeMix ----
        k_ln<<<NROWS, 256, 0, stream>>>(x, ln1_g + l * DC, ln1_b + l * DC, h);
        launch_gemm(stream, h, Wr, rbuf, NROWS, DC, DC, DC, DC, DC, 1, nullptr, 0, 0); // sigmoid
        launch_gemm(stream, h, Wk, kbuf, NROWS, DC, DC, DC, DC, DC, 0, nullptr, 0, 0);
        launch_gemm(stream, h, Wv, vbuf, NROWS, DC, DC, DC, DC, DC, 0, nullptr, 0, 0);
        if (l == 0) k_copy<<<ebN, 256, 0, stream>>>(vfirst, vbuf, NC);
        else        k_mix <<<ebN, 256, 0, stream>>>(vbuf, vfirst, NC);
        k_scan<<<(DB * DC + 255) / 256, 256, 0, stream>>>(kbuf, vbuf, att_decay + l * DC, state);
        k_mul<<<ebN, 256, 0, stream>>>(rsbuf, rbuf, state, NC);
        launch_gemm(stream, rsbuf, Wo, tmp, NROWS, DC, DC, DC, DC, DC, 0, nullptr, 0, 0);
        k_add<<<ebN, 256, 0, stream>>>(x, tmp, NC);

        // ---- routing ----
        k_ln<<<NROWS, 256, 0, stream>>>(x, ln2_g + l * DC, ln2_b + l * DC, h);
        k_routing<<<NROWS, 256, 0, stream>>>(h,
            conf_r + (size_t)l * DNR * DC, conf_t + (size_t)l * DNT * DC,
            critic_wd + (size_t)l * DC, critic_Wa + (size_t)l * DC * DE,
            capital + (size_t)l * DE, gate);

        // ---- bridge prefix ----
        k_add2<<<ebN, 256, 0, stream>>>(atmp, h, state, NC);
        launch_gemm(stream, atmp, Wdown, ubuf, NROWS, DR, DC, DC, DR, DR, 0, nullptr, 0, 0);
        for (int p = 0; p < DP; ++p)
            launch_gemm(stream, ubuf, Wup + (size_t)p * DR * DC, prefix + (size_t)p * DC,
                        NROWS, DC, DR, DR, DC, DP * DC, 0, nullptr, 0, 0);

        // ---- experts ----
        k_zero<<<ebN, 256, 0, stream>>>(finalb, NC);
        for (int e = 0; e < DNR; ++e) {
            launch_gemm(stream, h, eWk + ((size_t)l * DNR + e) * DC * DHID, hid,
                        NROWS, DHID, DC, DC, DHID, DHID, 2, nullptr, 0, 0);     // sqrelu
            launch_gemm(stream, hid, eWv + ((size_t)l * DNR + e) * DHID * DC, finalb,
                        NROWS, DC, DHID, DHID, DC, DC, 0, gate + e, DE, 1);     // gate*... +=
        }
        for (int j = 0; j < DNT; ++j) {
            const float* Wq  = tWq + ((size_t)l * DNT + j) * DC * DC;
            const float* Wkt = tWk + ((size_t)l * DNT + j) * DC * DC;
            const float* Wvt = tWv + ((size_t)l * DNT + j) * DC * DC;
            const float* Wot = tWo + ((size_t)l * DNT + j) * DC * DC;
            launch_gemm(stream, h, Wq, qbuf, NROWS, DC, DC, DC, DC, DC, 0, nullptr, 0, 0);
            launch_gemm(stream, prefix, Wkt, kp, NROWS * DP, DC, DC, DC, DC, DC, 0, nullptr, 0, 0);
            launch_gemm(stream, prefix, Wvt, vp, NROWS * DP, DC, DC, DC, DC, DC, 0, nullptr, 0, 0);
            k_attn<<<dim3(NROWS, DNH), DHD, 0, stream>>>(qbuf, kp, vp, tmp);
            launch_gemm(stream, tmp, Wot, finalb, NROWS, DC, DC, DC, DC, DC, 0,
                        gate + DNR + j, DE, 1);
        }
        k_add<<<ebN, 256, 0, stream>>>(x, finalb, NC);
    }

    // ---- final LN + head ----
    k_ln<<<NROWS, 256, 0, stream>>>(x, lno_g, lno_b, h);
    launch_gemm(stream, h, headW, (float*)d_out, NROWS, DV, DC, DC, DV, DV, 0, nullptr, 0, 0);
}
